// MHCrossAttn_38233798869263
// MI455X (gfx1250) — matrix operand order repaired and checked by reading, not compiled, not run on hardware
//
#include <hip/hip_runtime.h>
#include <hip/hip_bf16.h>

// ---------- types ----------
typedef __bf16 bf16_t;
typedef __attribute__((ext_vector_type(16))) __bf16 v16bf;
typedef __attribute__((ext_vector_type(8)))  __bf16 v8bf;
typedef __attribute__((ext_vector_type(8)))  float  v8f;

// D = A(16x32 bf16) * B(32x16 bf16) + C(16x16 f32)
static __device__ __forceinline__ v8f wmma_bf16(v16bf a, v16bf b, v8f c) {
  return __builtin_amdgcn_wmma_f32_16x16x32_bf16(false, a, false, b, (short)0, c, false, false);
}

// A-operand (16x32, 16-bit): lane row M=r; halves i = K = hi*16 + i (16 contiguous)
static __device__ __forceinline__ v16bf load_a(const bf16_t* base, int stride, int r, int hi) {
  const bf16_t* p = base + (size_t)r * stride + hi * 16;
  v8bf lo = *(const v8bf*)p;
  v8bf hh = *(const v8bf*)(p + 8);
  v16bf o;
#pragma unroll
  for (int i = 0; i < 8; ++i) { o[i] = lo[i]; o[i + 8] = hh[i]; }
  return o;
}

// B-operand (32x16, 16-bit): lane col N=r; halves i = K = hi*16 + i (16 contiguous)
static __device__ __forceinline__ v16bf load_b(const bf16_t* base, int stride, int r, int hi) {
  const bf16_t* p = base + (size_t)r * stride + hi * 16;
  v8bf lo = *(const v8bf*)p;
  v8bf hh = *(const v8bf*)(p + 8);
  v16bf o;
#pragma unroll
  for (int i = 0; i < 8; ++i) { o[i] = lo[i]; o[i + 8] = hh[i]; }
  return o;
}

// ---------- K0: weights fp32 -> bf16 (row-major kept) ----------
__global__ void k_cvt_w(const float* wq, const float* wk, const float* wv, const float* wp,
                        bf16_t* o) {
  int m = blockIdx.y;
  const float* s = (m == 0) ? wq : (m == 1) ? wk : (m == 2) ? wv : wp;
  int i = blockIdx.x * 256 + threadIdx.x;           // 0..65535
  o[(size_t)m * 65536 + i] = (bf16_t)s[i];
}

// ---------- K1: GroupNorm (32 groups x 8 ch x 1024 spatial), write bf16 position-major ----------
__global__ void k_groupnorm(const float* a, const float* b,
                            const float* gAw, const float* gAb,
                            const float* gBw, const float* gBb,
                            bf16_t* an, bf16_t* bn) {
  int which = blockIdx.y;
  const float* x  = which ? b   : a;
  const float* gw = which ? gBw : gAw;
  const float* gb = which ? gBb : gAb;
  bf16_t* o = which ? bn : an;
  int bb = blockIdx.x >> 5, g = blockIdx.x & 31;
  const float* xp = x + ((size_t)bb * 256 + g * 8) * 1024;
  int tid = threadIdx.x;
  float s = 0.f, s2 = 0.f;
  for (int i = tid; i < 8192; i += 256) { float v = xp[i]; s += v; s2 += v * v; }
  __shared__ float rs[256], rq[256];
  rs[tid] = s; rq[tid] = s2; __syncthreads();
  for (int off = 128; off > 0; off >>= 1) {
    if (tid < off) { rs[tid] += rs[tid + off]; rq[tid] += rq[tid + off]; }
    __syncthreads();
  }
  float mean = rs[0] * (1.f / 8192.f);
  float var  = rq[0] * (1.f / 8192.f) - mean * mean;
  float rstd = rsqrtf(var + 1e-6f);
  for (int i = tid; i < 8192; i += 256) {
    int c = i >> 10, n = i & 1023;
    float vv = (xp[i] - mean) * rstd * gw[g * 8 + c] + gb[g * 8 + c];
    o[((size_t)bb * 1024 + n) * 256 + g * 8 + c] = (bf16_t)vv;   // position-major
  }
}

// ---------- K2: Q/K/V projections. Q,K stored position-major; V channel-major ----------
__global__ void k_qkv(const bf16_t* an, const bf16_t* bn, const bf16_t* wbf,
                      const float* bq, const float* bk, const float* bv,
                      bf16_t* q, bf16_t* k, bf16_t* v) {
  int proj = blockIdx.y, bb = blockIdx.z;
  int wave = threadIdx.x >> 5, lane = threadIdx.x & 31;
  int r = lane & 15, hi = lane >> 4;
  int t = blockIdx.x * 8 + wave;                     // 0..1023
  int mt = t & 15, nt = t >> 4;                      // 16 m-tiles x 64 n-tiles
  const bf16_t* X = ((proj == 0) ? bn : an) + (size_t)bb * 1024 * 256;
  const bf16_t* W = wbf + (size_t)proj * 65536;
  const float* bias = (proj == 0) ? bq : (proj == 1) ? bk : bv;
  v8f acc;
#pragma unroll
  for (int i = 0; i < 8; ++i) acc[i] = bias[mt * 16 + 8 * hi + i];
  for (int kc = 0; kc < 256; kc += 32) {
    v16bf A = load_a(W + (size_t)mt * 16 * 256 + kc, 256, r, hi);
    v16bf B = load_b(X + (size_t)nt * 16 * 256 + kc, 256, r, hi);
    acc = wmma_bf16(A, B, acc);
  }
  if (proj < 2) {                                    // position-major: [(b,n),o]
    bf16_t* dst = ((proj == 0) ? q : k) + (size_t)bb * 1024 * 256;
    v8bf o;
#pragma unroll
    for (int i = 0; i < 8; ++i) o[i] = (bf16_t)acc[i];
    *(v8bf*)(dst + (size_t)(nt * 16 + r) * 256 + mt * 16 + 8 * hi) = o;
  } else {                                           // channel-major: [(b,o),n]
    bf16_t* dst = v + (size_t)bb * 256 * 1024;
#pragma unroll
    for (int i = 0; i < 8; ++i)
      dst[(size_t)(mt * 16 + 8 * hi + i) * 1024 + nt * 16 + r] = (bf16_t)acc[i];
  }
}

// ---------- K3: flash attention per (b,h), query tile = 16, key chunks of 32 ----------
__global__ void k_attn(const bf16_t* qm, const bf16_t* km, const bf16_t* vm, bf16_t* h) {
  int wave = threadIdx.x >> 5, lane = threadIdx.x & 31;
  int r = lane & 15, hi = lane >> 4;
  int id = blockIdx.x * 8 + wave;                    // 0..4095
  int qt = id & 63;                                  // query tile
  int bh = id >> 6;
  int bb = bh >> 2, hh = bh & 3;
  const bf16_t* qb = qm + (size_t)bb * 1024 * 256 + hh * 64;   // stride 256, d contiguous
  const bf16_t* kb = km + (size_t)bb * 1024 * 256 + hh * 64;
  const bf16_t* vb = vm + ((size_t)bb * 256 + hh * 64) * 1024; // stride 1024, pos contiguous

  // Q as B-operand (d x query), fixed for the whole key loop
  v16bf bq0 = load_b(qb + (size_t)qt * 16 * 256,      256, r, hi);  // d 0..31
  v16bf bq1 = load_b(qb + (size_t)qt * 16 * 256 + 32, 256, r, hi); // d 32..63

  v8f acc0 = {}, acc1 = {}, acc2 = {}, acc3 = {};    // out^T tiles (d x query)
  float m_run = -3.0e38f, l_run = 0.f;
  const float sc = 0.0625f;                          // 1/sqrt(256)

  for (int jc = 0; jc < 1024; jc += 32) {
    // S^T = K^T * Q : lane = query column, rows = keys
    v8f s0 = {}, s1 = {};
    s0 = wmma_bf16(load_a(kb + (size_t)jc * 256,            256, r, hi), bq0, s0);
    s0 = wmma_bf16(load_a(kb + (size_t)jc * 256 + 32,       256, r, hi), bq1, s0);
    s1 = wmma_bf16(load_a(kb + (size_t)(jc + 16) * 256,     256, r, hi), bq0, s1);
    s1 = wmma_bf16(load_a(kb + (size_t)(jc + 16) * 256 + 32,256, r, hi), bq1, s1);

    float cm = -3.0e38f;
#pragma unroll
    for (int i = 0; i < 8; ++i) {
      s0[i] *= sc; s1[i] *= sc;
      cm = fmaxf(cm, fmaxf(s0[i], s1[i]));
    }
    cm = fmaxf(cm, __shfl_xor(cm, 16, 32));          // other 16 keys (partner half)
    float mn = fmaxf(m_run, cm);
    float corr = __expf(m_run - mn);

    float e0[8], e1[8]; float rsum = 0.f;
#pragma unroll
    for (int i = 0; i < 8; ++i) {
      e0[i] = __expf(s0[i] - mn);
      e1[i] = __expf(s1[i] - mn);
      rsum += e0[i] + e1[i];
    }
    rsum += __shfl_xor(rsum, 16, 32);
    l_run = l_run * corr + rsum;
    m_run = mn;

    // Pack P^T into B operand (keys x query): lane needs keys hi*16+0..15 of this chunk.
    // Own regs hold keys {hi*8+v} (e0) and {16+hi*8+v} (e1); partner (xor 16) has the rest.
    v16bf P;
#pragma unroll
    for (int i = 0; i < 8; ++i) {
      float o0 = __shfl_xor(e0[i], 16, 32);
      float o1 = __shfl_xor(e1[i], 16, 32);
      P[i]     = (bf16_t)(hi ? o1 : e0[i]);
      P[i + 8] = (bf16_t)(hi ? e1[i] : o0);
    }
#pragma unroll
    for (int i = 0; i < 8; ++i) { acc0[i] *= corr; acc1[i] *= corr; acc2[i] *= corr; acc3[i] *= corr; }

    // out^T += V^T(16x32 keys) * P^T(32 keys x 16 queries), 4 d-tiles
    acc0 = wmma_bf16(load_a(vb +              jc, 1024, r, hi), P, acc0);
    acc1 = wmma_bf16(load_a(vb + 16 * 1024 + jc, 1024, r, hi), P, acc1);
    acc2 = wmma_bf16(load_a(vb + 32 * 1024 + jc, 1024, r, hi), P, acc2);
    acc3 = wmma_bf16(load_a(vb + 48 * 1024 + jc, 1024, r, hi), P, acc3);
  }

  float inv = 1.0f / l_run;
  // store h position-major: [(b, n), c] with c = hh*64 + dtile*16 + 8*hi + v (contiguous run of 8)
  bf16_t* hp = h + ((size_t)bb * 1024 + qt * 16 + r) * 256 + hh * 64 + hi * 8;
  v8bf o;
#pragma unroll
  for (int i = 0; i < 8; ++i) o[i] = (bf16_t)(acc0[i] * inv);
  *(v8bf*)(hp + 0) = o;
#pragma unroll
  for (int i = 0; i < 8; ++i) o[i] = (bf16_t)(acc1[i] * inv);
  *(v8bf*)(hp + 16) = o;
#pragma unroll
  for (int i = 0; i < 8; ++i) o[i] = (bf16_t)(acc2[i] * inv);
  *(v8bf*)(hp + 32) = o;
#pragma unroll
  for (int i = 0; i < 8; ++i) o[i] = (bf16_t)(acc3[i] * inv);
  *(v8bf*)(hp + 48) = o;
}

// ---------- K4: out = (skip_b + Wp*h + bp) / sqrt(2), fp32 output (B,C,H,W) ----------
__global__ void k_out(const bf16_t* h, const bf16_t* wp, const float* bp,
                      const float* skip, float* out) {
  int bb = blockIdx.z;
  int wave = threadIdx.x >> 5, lane = threadIdx.x & 31;
  int r = lane & 15, hi = lane >> 4;
  int t = blockIdx.x * 8 + wave;
  int mt = t & 15, nt = t >> 4;
  const bf16_t* X = h + (size_t)bb * 1024 * 256;
  v8f acc;
#pragma unroll
  for (int i = 0; i < 8; ++i) acc[i] = bp[mt * 16 + 8 * hi + i];
  for (int kc = 0; kc < 256; kc += 32) {
    v16bf A = load_a(wp + (size_t)mt * 16 * 256 + kc, 256, r, hi);
    v16bf B = load_b(X  + (size_t)nt * 16 * 256 + kc, 256, r, hi);
    acc = wmma_bf16(A, B, acc);
  }
  const float is2 = 0.70710678118654752f;
#pragma unroll
  for (int i = 0; i < 8; ++i) {
    int m = mt * 16 + 8 * hi + i;
    int n = nt * 16 + r;
    size_t idx = ((size_t)bb * 256 + m) * 1024 + n;
    out[idx] = (skip[idx] + acc[i]) * is2;
  }
}

extern "C" void kernel_launch(void* const* d_in, const int* in_sizes, int n_in,
                              void* d_out, int out_size, void* d_ws, size_t ws_size,
                              hipStream_t stream) {
  const float* a   = (const float*)d_in[0];
  const float* b   = (const float*)d_in[1];
  const float* Wq  = (const float*)d_in[2];
  const float* bq  = (const float*)d_in[3];
  const float* Wk  = (const float*)d_in[4];
  const float* bk  = (const float*)d_in[5];
  const float* Wv  = (const float*)d_in[6];
  const float* bv  = (const float*)d_in[7];
  const float* Wp  = (const float*)d_in[8];
  const float* bp  = (const float*)d_in[9];
  const float* gAw = (const float*)d_in[10];
  const float* gAb = (const float*)d_in[11];
  const float* gBw = (const float*)d_in[12];
  const float* gBb = (const float*)d_in[13];
  float* out = (float*)d_out;

  const size_t ACT = (size_t)16 * 1024 * 256;        // elements per activation tensor
  bf16_t* an  = (bf16_t*)d_ws;                       // normalized a, position-major
  bf16_t* bn  = an + ACT;                            // normalized b, position-major
  bf16_t* qw  = bn + ACT;                            // Q position-major
  bf16_t* kw  = qw + ACT;                            // K position-major
  bf16_t* vw  = kw + ACT;                            // V channel-major
  bf16_t* hw  = vw + ACT;                            // attention out, position-major
  bf16_t* wbf = hw + ACT;                            // 4 weight matrices bf16

  k_cvt_w    <<<dim3(256, 4),     256, 0, stream>>>(Wq, Wk, Wv, Wp, wbf);
  k_groupnorm<<<dim3(512, 2),     256, 0, stream>>>(a, b, gAw, gAb, gBw, gBb, an, bn);
  k_qkv      <<<dim3(128, 3, 16), 256, 0, stream>>>(an, bn, wbf, bq, bk, bv, qw, kw, vw);
  k_attn     <<<dim3(512),        256, 0, stream>>>(qw, kw, vw, hw);
  k_out      <<<dim3(128, 1, 16), 256, 0, stream>>>(hw, wbf + 3 * 65536, bp, b, out);
}
